// LexerBPE_858993459868
// MI455X (gfx1250) — matrix operand order repaired and checked by reading, not compiled, not run on hardware
//
#include <hip/hip_runtime.h>
#include <math.h>

// ---- problem geometry (from reference) ----
#define D_BPE 1024          // K
#define D_WORD 256          // N
#define KI_STEPS 32         // K / 32
#define NT 16               // N / 16
#define FRAG_ELEMS (NT * 32 * 16)            // bf16 elems per k-step buffer = 8192
#define FRAG_BYTES (FRAG_ELEMS * 2)          // 16 KB

typedef __attribute__((ext_vector_type(16))) __bf16 v16bf;
typedef __attribute__((ext_vector_type(8)))  float  v8f;
typedef __attribute__((ext_vector_type(4)))  float  v4f;

// ---------------------------------------------------------------------------
// K0: pack W (fp32, [256,1024] row-major) into bf16 per-lane WMMA B fragments.
// Layout: wsB[ ((ki*NT + nt)*32 + lane)*16 + j ] = bf16( W[nt*16 + (lane&15)]
//                                                        [ki*32 + (lane<16?0:16) + j] )
// (B 32x16 bf16: lane = column N, 16 consecutive K per lane, K-halves split at lane 16)
// ---------------------------------------------------------------------------
__global__ void pack_W_kernel(const float* __restrict__ W, __bf16* __restrict__ wsB) {
    int id   = blockIdx.x * 256 + threadIdx.x;      // 0 .. 262143
    int j    =  id        & 15;
    int lane = (id >> 4)  & 31;
    int nt   = (id >> 9)  & 15;
    int ki   =  id >> 13;
    int n = nt * 16 + (lane & 15);
    int k = ki * 32 + ((lane < 16) ? 0 : 16) + j;
    wsB[id] = (__bf16)W[n * D_BPE + k];
}

// ---------------------------------------------------------------------------
// K1: word start offsets via binary search over sorted seg_ids.
// starts[w] = lower_bound(seg_ids, w); starts[num_words] = T.
// ---------------------------------------------------------------------------
__global__ void seg_starts_kernel(const int* __restrict__ seg, int* __restrict__ starts,
                                  int T, int num_words) {
    int w = blockIdx.x * 256 + threadIdx.x;
    if (w > num_words) return;
    int lo = 0, hi = T;
    while (lo < hi) {
        int mid = (lo + hi) >> 1;
        if (seg[mid] < w) lo = mid + 1; else hi = mid;
    }
    starts[w] = lo;
}

// ---------------------------------------------------------------------------
// K2: h = tanh(x @ W^T + b), bf16 WMMA with fp32 accumulate.
// Block = 256 threads (8 waves). Wave w: rows [blk*128 + w*16, +16), all 256 cols.
// W fragments stream through LDS via async-load, double buffered.
// ---------------------------------------------------------------------------
__global__ __launch_bounds__(256) void gemm_tanh_kernel(
    const float* __restrict__ x, const __bf16* __restrict__ wsB,
    const float* __restrict__ bias, float* __restrict__ h) {

    __shared__ __bf16 smem[2][FRAG_ELEMS];   // 2 x 16 KB

    const int tid  = threadIdx.x;
    const int lane = tid & 31;
    const int wave = tid >> 5;
    const int rowBase = blockIdx.x * 128 + wave * 16;

    const unsigned smem0 = (unsigned)(unsigned long long)(void*)&smem[0][0];

    v8f acc[NT];
#pragma unroll
    for (int nt = 0; nt < NT; ++nt) {
        v8f z = {0.f, 0.f, 0.f, 0.f, 0.f, 0.f, 0.f, 0.f};
        acc[nt] = z;
    }

    const int rowA   = rowBase + (lane & 15);
    const float* xrow = x + (size_t)rowA * D_BPE;
    const int ksel   = (lane < 16) ? 0 : 16;

    // -- prologue: stage k-step 0 into buffer 0 (4 x b128 per lane = 16 KB/WG) --
    {
        unsigned long long src = (unsigned long long)wsB;
#pragma unroll
        for (int c = 0; c < 4; ++c) {
            unsigned off  = (unsigned)tid * 16u + (unsigned)c * 4096u;
            unsigned ldsa = smem0 + off;
            asm volatile("global_load_async_to_lds_b128 %0, %1, %2"
                         :: "v"(ldsa), "v"(off), "s"(src) : "memory");
        }
    }

    for (int ki = 0; ki < KI_STEPS; ++ki) {
        // issue next buffer, then wait for the current one (in-order completion)
        if (ki + 1 < KI_STEPS) {
            unsigned long long src =
                (unsigned long long)(wsB + (size_t)(ki + 1) * FRAG_ELEMS);
            unsigned bufo = ((ki + 1) & 1) ? (unsigned)FRAG_BYTES : 0u;
#pragma unroll
            for (int c = 0; c < 4; ++c) {
                unsigned off  = (unsigned)tid * 16u + (unsigned)c * 4096u;
                unsigned ldsa = smem0 + bufo + off;
                asm volatile("global_load_async_to_lds_b128 %0, %1, %2"
                             :: "v"(ldsa), "v"(off), "s"(src) : "memory");
            }
            asm volatile("s_wait_asynccnt 0x4" ::: "memory");
        } else {
            asm volatile("s_wait_asynccnt 0x0" ::: "memory");
        }
        __syncthreads();

        // ---- A fragment: 16 fp32 loads -> bf16, contiguous K order (k = 16*half + s) ----
        const int kk = ki * 32;
        v4f f0 = *(const v4f*)(xrow + kk + ksel);
        v4f f1 = *(const v4f*)(xrow + kk + ksel + 4);
        v4f f2 = *(const v4f*)(xrow + kk + ksel + 8);
        v4f f3 = *(const v4f*)(xrow + kk + ksel + 12);
        v16bf a;
        a[0]  = (__bf16)f0[0]; a[1]  = (__bf16)f0[1]; a[2]  = (__bf16)f0[2]; a[3]  = (__bf16)f0[3];
        a[4]  = (__bf16)f1[0]; a[5]  = (__bf16)f1[1]; a[6]  = (__bf16)f1[2]; a[7]  = (__bf16)f1[3];
        a[8]  = (__bf16)f2[0]; a[9]  = (__bf16)f2[1]; a[10] = (__bf16)f2[2]; a[11] = (__bf16)f2[3];
        a[12] = (__bf16)f3[0]; a[13] = (__bf16)f3[1]; a[14] = (__bf16)f3[2]; a[15] = (__bf16)f3[3];

        // ---- 16 WMMAs over the full N strip, B fragments from LDS ----
        const __bf16* bufp = &smem[ki & 1][0];
#pragma unroll
        for (int nt = 0; nt < NT; ++nt) {
            v16bf bfrag = *(const v16bf*)(bufp + (nt * 32 + lane) * 16);
            acc[nt] = __builtin_amdgcn_wmma_f32_16x16x32_bf16(
                /*neg_a=*/false, a, /*neg_b=*/false, bfrag,
                /*c_mod=*/(short)0, acc[nt], /*reuse_a=*/false, /*reuse_b=*/false);
        }
        __syncthreads();   // protect buffer (ki&1) before it is re-staged
    }

    // ---- epilogue: bias + tanh + store h [T, 256] ----
    // C/D layout: VGPR v, lanes 0-15 -> M=v; lanes 16-31 -> M=v+8; N = lane&15
    const int colBase = lane & 15;
    const int rowOff  = (lane < 16) ? 0 : 8;
#pragma unroll
    for (int nt = 0; nt < NT; ++nt) {
        float bv = bias[nt * 16 + colBase];
#pragma unroll
        for (int v = 0; v < 8; ++v) {
            float val = tanhf(acc[nt][v] + bv);
            int row = rowBase + rowOff + v;
            h[(size_t)row * D_WORD + nt * 16 + colBase] = val;
        }
    }
}

// ---------------------------------------------------------------------------
// K3: deterministic ragged segment-sum. One block per word, one thread per col.
// Empty words naturally produce 0 (matches segment_sum semantics).
// ---------------------------------------------------------------------------
__global__ void seg_reduce_kernel(const float* __restrict__ h,
                                  const int* __restrict__ starts,
                                  float* __restrict__ out) {
    int w = blockIdx.x;
    int c = threadIdx.x;
    int s = starts[w];
    int e = starts[w + 1];
    float sum = 0.f;
    for (int t = s; t < e; ++t) sum += h[(size_t)t * D_WORD + c];
    out[(size_t)w * D_WORD + c] = sum;
}

// ---------------------------------------------------------------------------
extern "C" void kernel_launch(void* const* d_in, const int* in_sizes, int n_in,
                              void* d_out, int out_size, void* d_ws, size_t ws_size,
                              hipStream_t stream) {
    const float* x   = (const float*)d_in[0];   // [T, 1024]
    const float* W   = (const float*)d_in[1];   // [256, 1024]
    const float* b   = (const float*)d_in[2];   // [256]
    const int*   seg = (const int*)d_in[3];     // [T], sorted
    const int T         = in_sizes[3];
    const int num_words = out_size / D_WORD;

    // workspace layout: [0,512K) packed W bf16; [512K,512K+~128K) starts; [1M, 1M+64M) h
    __bf16* wsB   = (__bf16*)d_ws;
    int*    starts = (int*)((char*)d_ws + (512u << 10));
    float*  h     = (float*)((char*)d_ws + (1u << 20));

    pack_W_kernel<<<(D_WORD * D_BPE) / 256, 256, 0, stream>>>(W, wsB);
    seg_starts_kernel<<<(num_words + 1 + 255) / 256, 256, 0, stream>>>(seg, starts, T, num_words);
    gemm_tanh_kernel<<<T / 128, 256, 0, stream>>>(x, wsB, b, h);
    seg_reduce_kernel<<<num_words, 256, 0, stream>>>(h, starts, (float*)d_out);
}
